// RNN_13855564496941
// MI455X (gfx1250) — matrix operand order repaired and checked by reading, not compiled, not run on hardware
//
#include <hip/hip_runtime.h>

#define T_STEPS 1024
#define BATCH   4096
#define H_DIM   50
#define DT      0.1f
#define COLSTR  72   // padded LDS column stride in bf16 elems (144B, 16B-aligned, conflict-free)

typedef __attribute__((ext_vector_type(16))) __bf16 v16bf;
typedef __attribute__((ext_vector_type(8)))  __bf16 v8bf;
typedef __attribute__((ext_vector_type(8)))  float  v8f;

__device__ __forceinline__ float act_1p_tanh(float h) {
    // 1 + tanh(h) = 2 / (1 + exp(-2h)) ; exp(-2h) = exp2(h * -2*log2(e))
    float e = __builtin_amdgcn_exp2f(h * -2.8853900817779268f);
    return 2.0f * __builtin_amdgcn_rcpf(1.0f + e);
}

__global__ __launch_bounds__(128) void rnn_scan_wmma(
    const float* __restrict__ x,     // (T, B)
    const float* __restrict__ Jin,   // (H)
    const float* __restrict__ Jrec,  // (H, H) row-major
    const float* __restrict__ Jout,  // (H)
    const float* __restrict__ bias,  // (H)
    const float* __restrict__ h0,    // (H, B)
    float* __restrict__ out)         // (T, B)
{
    __shared__ __bf16 aExt[2][16][COLSTR];  // double-buffered activation, column-major per col
    __shared__ float  outP[2][4][16];       // double-buffered per-wave output partials

    const int tid     = threadIdx.x;
    const int w       = tid >> 5;        // wave id 0..3 -> M row-tile
    const int lane    = tid & 31;
    const int col16   = lane & 15;       // column within tile / matrix row for A operand
    const int hi      = lane >> 4;       // lane half
    const int col     = blockIdx.x * 16 + col16;
    const int rowBase = 16 * w + 8 * hi; // first hidden row held in this lane's accumulators

    // ---- per-lane row constants + initial hidden state (WMMA C/D register layout) ----
    float h[8], cj[8], cb[8], jo[8];
    #pragma unroll
    for (int r = 0; r < 8; ++r) {
        const int row = rowBase + r;
        const bool ok = row < H_DIM;
        cj[r] = ok ? DT * Jin[row]  : 0.0f;
        cb[r] = ok ? DT * bias[row] : 0.0f;
        jo[r] = ok ? Jout[row]      : 0.0f;
        h[r]  = ok ? h0[(size_t)row * BATCH + col] : 0.0f;
    }

    // ---- build Jrec A operands once (bf16, WMMA 16x32 A striping), zero-padded to 64x64 ----
    // lanes 0-15 row M=lane: elems j -> K=j ; lanes 16-31 same rows, K += 16
    const int arow = 16 * w + col16;
    v16bf A0, A1;
    #pragma unroll
    for (int j = 0; j < 16; ++j) {
        const int k0 = j + 16 * hi;   // K within 0..31
        const int k1 = k0 + 32;                            // K within 32..63
        float v0 = (arow < H_DIM && k0 < H_DIM) ? Jrec[arow * H_DIM + k0] : 0.0f;
        float v1 = (arow < H_DIM && k1 < H_DIM) ? Jrec[arow * H_DIM + k1] : 0.0f;
        A0[j] = (__bf16)v0;
        A1[j] = (__bf16)v1;
    }

    // ---- seed activation buffer 0 with a(h0) for this wave's rows ----
    {
        v8bf av;
        #pragma unroll
        for (int r = 0; r < 8; ++r) av[r] = (__bf16)act_1p_tanh(h[r]);
        *(v8bf*)&aExt[0][col16][rowBase] = av;   // 16B aligned, contiguous 8 rows
    }

    float xt = x[col];  // x_0 for this column

    for (int t = 0; t < T_STEPS; ++t) {
        const int p = t & 1;
        __syncthreads();   // aExt[p] / outP[p] written last step are now visible

        // wave 0 drains the previous step's output (pipelined one step behind)
        if (t > 0 && w == 0 && hi == 0) {
            float s = outP[p][0][col16] + outP[p][1][col16]
                    + outP[p][2][col16] + outP[p][3][col16];
            out[(size_t)(t - 1) * BATCH + col] = s;
        }

        // prefetch / preload future inputs
        if (t + 8 < T_STEPS)
            __builtin_prefetch(&x[(size_t)(t + 8) * BATCH + col], 0, 0);
        const float xn = (t + 1 < T_STEPS) ? x[(size_t)(t + 1) * BATCH + col] : 0.0f;

        // ---- B operands: lane reads contiguous K-runs of its column from LDS ----
        const __bf16* bp = &aExt[p][col16][16 * hi];
        v8bf b00 = *(const v8bf*)(bp);        // K = 16*hi + 0..7
        v8bf b01 = *(const v8bf*)(bp + 8);    // K = 16*hi + 8..15
        v8bf b10 = *(const v8bf*)(bp + 32);   // K = 32 + 16*hi + 0..7
        v8bf b11 = *(const v8bf*)(bp + 40);
        v16bf B0 = __builtin_shufflevector(b00, b01, 0,1,2,3,4,5,6,7,8,9,10,11,12,13,14,15);
        v16bf B1 = __builtin_shufflevector(b10, b11, 0,1,2,3,4,5,6,7,8,9,10,11,12,13,14,15);

        // ---- D = Jrec_tile @ a  (16x16x64 via two 16x16x32 bf16 WMMAs, f32 accum) ----
        v8f d = {0.f,0.f,0.f,0.f,0.f,0.f,0.f,0.f};
        d = __builtin_amdgcn_wmma_f32_16x16x32_bf16(false, A0, false, B0, (short)0, d, false, false);
        d = __builtin_amdgcn_wmma_f32_16x16x32_bf16(false, A1, false, B1, (short)0, d, false, false);

        // ---- Euler update, output partial, next activation ----
        float part = 0.0f;
        v8bf av;
        #pragma unroll
        for (int r = 0; r < 8; ++r) {
            float hn = (1.0f - DT) * h[r] + DT * d[r] + cj[r] * xt + cb[r];
            h[r] = hn;
            part = fmaf(jo[r], hn, part);
            av[r] = (__bf16)act_1p_tanh(hn);
        }
        part += __shfl_xor(part, 16, 32);   // combine lane-pair halves of each column

        const int q = p ^ 1;
        *(v8bf*)&aExt[q][col16][rowBase] = av;
        if (hi == 0) outP[q][w][col16] = part;

        xt = xn;
    }

    // drain final step's output (it sits in buffer parity T&1 == 0)
    __syncthreads();
    if (w == 0 && hi == 0) {
        const int p = T_STEPS & 1;
        float s = outP[p][0][col16] + outP[p][1][col16]
                + outP[p][2][col16] + outP[p][3][col16];
        out[(size_t)(T_STEPS - 1) * BATCH + col] = s;
    }
}

extern "C" void kernel_launch(void* const* d_in, const int* in_sizes, int n_in,
                              void* d_out, int out_size, void* d_ws, size_t ws_size,
                              hipStream_t stream) {
    const float* x    = (const float*)d_in[0];  // inpt_data (T,B)
    const float* Jin  = (const float*)d_in[1];  // (H,1) flat
    const float* Jrec = (const float*)d_in[2];  // (H,H)
    const float* Jout = (const float*)d_in[3];  // (1,H) flat
    const float* bias = (const float*)d_in[4];  // (H,1) flat
    const float* h0   = (const float*)d_in[5];  // (H,B)
    float* out = (float*)d_out;                 // (T,B)

    dim3 grid(BATCH / 16);   // 256 blocks, each owns 16 batch columns for all T steps
    dim3 block(128);         // 4 waves: one 16-row tile of the hidden state each
    rnn_scan_wmma<<<grid, block, 0, stream>>>(x, Jin, Jrec, Jout, bias, h0, out);
}
